// DCGRUCell_34583076668042
// MI455X (gfx1250) — matrix operand order repaired and checked by reading, not compiled, not run on hardware
//
#include <hip/hip_runtime.h>
#include <hip/hip_bf16.h>

typedef __attribute__((ext_vector_type(16))) __bf16 v16bf;
typedef __attribute__((ext_vector_type(8)))  __bf16 v8bf;
typedef __attribute__((ext_vector_type(8)))  float  v8f;

#define NODES 8192
#define BATCH 16
#define INDIM 2
#define HID   64
#define FDIM  66            // INDIM + HID
#define NMAT  5
#define KDIM  330           // FDIM * NMAT
#define KPAD  352           // 11 * 32
#define EDGES 131072
#define ROWS  (BATCH * NODES)   // 131072
#define FB    (FDIM * BATCH)    // 1056

// ---------------- elementwise / staging kernels ----------------

__global__ void k_zero_f32(float* p, int n4) {
    int i = blockIdx.x * blockDim.x + threadIdx.x;
    if (i < n4) ((float4*)p)[i] = make_float4(0.f, 0.f, 0.f, 0.f);
}

__global__ void k_zero_padA(__bf16* A) {
    int i = blockIdx.x * blockDim.x + threadIdx.x;   // ROWS * (KPAD-KDIM)
    if (i >= ROWS * (KPAD - KDIM)) return;
    int row = i / (KPAD - KDIM);
    int k   = KDIM + i % (KPAD - KDIM);
    A[(size_t)row * KPAD + k] = (__bf16)0.f;
}

// Wt[o, k'] = W[f*NMAT+m, o] with k' = m*FDIM+f (bf16, zero-padded K)
__global__ void k_prep_w(const float* __restrict__ Wg, const float* __restrict__ Wc,
                         __bf16* __restrict__ Wtg, __bf16* __restrict__ Wtc) {
    int i = blockIdx.x * blockDim.x + threadIdx.x;   // 192 * KPAD
    if (i >= 192 * KPAD) return;
    int o = i / KPAD, k = i % KPAD;
    float v = 0.f;
    if (k < KDIM) {
        int m = k / FDIM, f = k % FDIM;
        v = (o < 128) ? Wg[(f * NMAT + m) * 128 + o]
                      : Wc[(f * NMAT + m) * HID + (o - 128)];
    }
    if (o < 128) Wtg[o * KPAD + k] = (__bf16)v;
    else         Wtc[(o - 128) * KPAD + k] = (__bf16)v;
}

// x0[n, f, b] = concat(inputs, state)
__global__ void k_build_x0(const float* __restrict__ inp, const float* __restrict__ st,
                           float* __restrict__ x0) {
    int i = blockIdx.x * blockDim.x + threadIdx.x;   // NODES*FB, i = (n*FDIM+f)*16+b
    if (i >= NODES * FB) return;
    int b = i & 15, fn = i >> 4;
    int f = fn % FDIM, n = fn / FDIM;
    x0[i] = (f < INDIM) ? inp[b * (NODES * INDIM) + n * INDIM + f]
                        : st [b * (NODES * HID)   + n * HID   + (f - INDIM)];
}

// scatter SpMM: y[rows[e]] += vals[e] * x[cols[e]]  (row length FB, float4/thread)
__global__ void k_spmm(const float* __restrict__ vals, const int* __restrict__ rows,
                       const int* __restrict__ cols, const float* __restrict__ x,
                       float* __restrict__ y) {
    int i = blockIdx.x * blockDim.x + threadIdx.x;   // EDGES * FB/4
    if (i >= EDGES * (FB / 4)) return;
    int e = i / (FB / 4), q = i % (FB / 4);
    float v = vals[e];
    int r = rows[e], c = cols[e];
    float4 xv = ((const float4*)(x + (size_t)c * FB))[q];
    float* yp = y + (size_t)r * FB + q * 4;
    unsafeAtomicAdd(yp + 0, v * xv.x);
    unsafeAtomicAdd(yp + 1, v * xv.y);
    unsafeAtomicAdd(yp + 2, v * xv.z);
    unsafeAtomicAdd(yp + 3, v * xv.w);
}

// A[b*N+n, slot*FDIM+f] = bf16(y[n,f,b])
__global__ void k_conv_slot(const float* __restrict__ y, __bf16* __restrict__ A, int slot) {
    int i = blockIdx.x * blockDim.x + threadIdx.x;
    if (i >= NODES * FB) return;
    int b = i & 15, fn = i >> 4;
    int f = fn % FDIM, n = fn / FDIM;
    A[(size_t)(b * NODES + n) * KPAD + slot * FDIM + f] = (__bf16)y[i];
}

// Chebyshev: A slot = bf16(2*y2 - x0)
__global__ void k_conv_cheb(const float* __restrict__ y2, const float* __restrict__ x0,
                            __bf16* __restrict__ A, int slot) {
    int i = blockIdx.x * blockDim.x + threadIdx.x;
    if (i >= NODES * FB) return;
    int b = i & 15, fn = i >> 4;
    int f = fn % FDIM, n = fn / FDIM;
    A[(size_t)(b * NODES + n) * KPAD + slot * FDIM + f] = (__bf16)(2.f * y2[i] - x0[i]);
}

// ---------------- WMMA GEMM with fused GRU epilogues ----------------
// C[row, col] = A[row, :] . Wt[col, :] + bias[col];  block tile 64x32, wave tile 16x16.
__global__ __launch_bounds__(256) void k_wmma_gemm(
        const __bf16* __restrict__ A, const __bf16* __restrict__ Wt,
        const float* __restrict__ bias, const float* __restrict__ state,
        float* __restrict__ ubuf, float* __restrict__ x0f, float* __restrict__ out,
        int mode) {
    int lane = threadIdx.x & 31;
    int wave = threadIdx.x >> 5;
    int row0 = blockIdx.x * 64 + (wave & 3) * 16;
    int col0 = blockIdx.y * 32 + (wave >> 2) * 16;

    // A fragment: lane holds row M=lane&15; elems 0..7 -> k = kb..kb+7, 8..15 -> kb+16..kb+23, kb = 8*(lane>=16)
    const __bf16* Ap = A  + (size_t)(row0 + (lane & 15)) * KPAD + ((lane >> 4) << 3);
    // B fragment: lane holds col N=lane&15; elems 0..7 -> k = kb..kb+7, 8..15 -> kb+16..kb+23, kb = 8*(lane>=16)
    const __bf16* Bp = Wt + (size_t)(col0 + (lane & 15)) * KPAD + ((lane >> 4) << 3);

    v8f acc = {};
    for (int kt = 0; kt < KPAD; kt += 32) {
        union { v16bf v; v8bf h[2]; } a, b;
        a.h[0] = *(const v8bf*)(Ap + kt);
        a.h[1] = *(const v8bf*)(Ap + kt + 16);
        b.h[0] = *(const v8bf*)(Bp + kt);
        b.h[1] = *(const v8bf*)(Bp + kt + 16);
        acc = __builtin_amdgcn_wmma_f32_16x16x32_bf16(
                  false, a.v, false, b.v, (short)0, acc, false, false);
    }

    int col = col0 + (lane & 15);
    float bv = bias[col];
#pragma unroll
    for (int r = 0; r < 8; r++) {
        int row = row0 + r + ((lane >> 4) << 3);   // D: VGPR r -> M=r (lanes<16) / r+8
        int b   = row >> 13;                       // row / NODES
        int n   = row & (NODES - 1);
        float v = acc[r] + bv;
        if (mode == 0) {                           // gate: sigmoid, split r/u
            float g = 1.f / (1.f + __expf(-v));
            if (col < HID) {                       // r-gate: write r*state into x0 for conv 2
                float rs = g * state[(size_t)b * (NODES * HID) + n * HID + col];
                x0f[((n * FDIM + (INDIM + col)) << 4) + b] = rs;
            } else {                               // u-gate
                ubuf[(size_t)row * HID + (col - HID)] = g;
            }
        } else {                                   // candidate: tanh + GRU blend
            float c = tanhf(v);
            float u = ubuf[(size_t)row * HID + col];
            float s = state[(size_t)b * (NODES * HID) + n * HID + col];
            float ns = u * s + (1.f - u) * c;
            size_t oi = (size_t)b * (NODES * HID) + n * HID + col;
            out[oi] = ns;
            out[(size_t)BATCH * NODES * HID + oi] = ns;   // tuple: two copies
        }
    }
}

// ---------------- host orchestration ----------------

extern "C" void kernel_launch(void* const* d_in, const int* in_sizes, int n_in,
                              void* d_out, int out_size, void* d_ws, size_t ws_size,
                              hipStream_t stream) {
    (void)in_sizes; (void)n_in; (void)out_size; (void)ws_size;
    const float* inputs = (const float*)d_in[0];
    const float* state  = (const float*)d_in[1];
    const int*   esrc   = (const int*)d_in[2];
    const int*   edst   = (const int*)d_in[3];
    const float* vals1  = (const float*)d_in[4];
    const float* vals2  = (const float*)d_in[5];
    const float* Wg     = (const float*)d_in[6];
    const float* bg     = (const float*)d_in[7];
    const float* Wc     = (const float*)d_in[8];
    const float* bc     = (const float*)d_in[9];
    float* out = (float*)d_out;

    char* ws = (char*)d_ws;
    size_t off = 0;
    auto alloc = [&](size_t bytes) {
        void* p = ws + off;
        off = (off + bytes + 255) & ~(size_t)255;
        return p;
    };
    __bf16* Abuf = (__bf16*)alloc((size_t)ROWS * KPAD * 2);   // 92.3 MB
    float*  x0f  = (float*)alloc((size_t)NODES * FB * 4);     // 34.6 MB
    float*  y1   = (float*)alloc((size_t)NODES * FB * 4);
    float*  y2   = (float*)alloc((size_t)NODES * FB * 4);
    float*  ubuf = (float*)alloc((size_t)ROWS * HID * 4);     // 33.5 MB
    __bf16* Wtg  = (__bf16*)alloc((size_t)128 * KPAD * 2);
    __bf16* Wtc  = (__bf16*)alloc((size_t)HID * KPAD * 2);

    dim3 blk(256);
    auto g1 = [](int n) { return dim3((unsigned)((n + 255) / 256)); };
    const int NFB = NODES * FB;

    k_zero_padA<<<g1(ROWS * (KPAD - KDIM)), blk, 0, stream>>>(Abuf);
    k_prep_w<<<g1(192 * KPAD), blk, 0, stream>>>(Wg, Wc, Wtg, Wtc);
    k_build_x0<<<g1(NFB), blk, 0, stream>>>(inputs, state, x0f);
    k_conv_slot<<<g1(NFB), blk, 0, stream>>>(x0f, Abuf, 0);

    auto diff_pass = [&]() {
        for (int m = 0; m < 2; m++) {
            const float* v  = (m == 0) ? vals1 : vals2;
            const int*   rr = (m == 0) ? esrc  : edst;
            const int*   cc = (m == 0) ? edst  : esrc;
            k_zero_f32<<<g1(NFB / 4), blk, 0, stream>>>(y1, NFB / 4);
            k_spmm<<<g1(EDGES * (FB / 4)), blk, 0, stream>>>(v, rr, cc, x0f, y1);
            k_conv_slot<<<g1(NFB), blk, 0, stream>>>(y1, Abuf, 1 + 2 * m);
            k_zero_f32<<<g1(NFB / 4), blk, 0, stream>>>(y2, NFB / 4);
            k_spmm<<<g1(EDGES * (FB / 4)), blk, 0, stream>>>(v, rr, cc, y1, y2);
            k_conv_cheb<<<g1(NFB), blk, 0, stream>>>(y2, x0f, Abuf, 2 + 2 * m);
        }
    };

    diff_pass();
    // gate GEMM: 128 output cols; writes r*state into x0f (state part) and u into ubuf
    k_wmma_gemm<<<dim3(ROWS / 64, 128 / 32), blk, 0, stream>>>(
        Abuf, Wtg, bg, state, ubuf, x0f, out, 0);
    // refresh A slot 0 with xc = concat(input, r*state)
    k_conv_slot<<<g1(NFB), blk, 0, stream>>>(x0f, Abuf, 0);
    diff_pass();
    // candidate GEMM: 64 output cols; fused tanh + GRU blend + dual store
    k_wmma_gemm<<<dim3(ROWS / 64, 64 / 32), blk, 0, stream>>>(
        Abuf, Wtc, bc, state, ubuf, x0f, out, 1);
}
